// Attention_21775484191158
// MI455X (gfx1250) — matrix operand order repaired and checked by reading, not compiled, not run on hardware
//
#include <hip/hip_runtime.h>
#include <math.h>

#define DIM 4096
#define NH  32
#define NKV 8
#define HD  128
#define SEQ 2048

typedef __attribute__((ext_vector_type(16))) unsigned short u16x16;
typedef __attribute__((ext_vector_type(8)))  unsigned short u16x8;
typedef __attribute__((ext_vector_type(16))) __bf16         v16bf;
typedef __attribute__((ext_vector_type(8)))  float          v8f;
typedef int v4i_vs __attribute__((vector_size(16)));   // matches async-LDS builtin param

__device__ __forceinline__ unsigned short f2bf(float f) {
    unsigned int u = __float_as_uint(f);
    u += 0x7FFFu + ((u >> 16) & 1u);   // round-to-nearest-even
    return (unsigned short)(u >> 16);
}

__device__ __forceinline__ v16bf bfvec(u16x16 u) {
    return __builtin_bit_cast(v16bf, u);
}

// ---- gfx1250 async copy global->LDS (ASYNCcnt), with sync fallback ----------
#if __has_builtin(__builtin_amdgcn_global_load_async_to_lds_b128)
#define ASYNC_LDS 1
#else
#define ASYNC_LDS 0
#endif

__device__ __forceinline__ void async_cp16(const unsigned short* g, unsigned short* l) {
#if ASYNC_LDS
    __builtin_amdgcn_global_load_async_to_lds_b128(
        (__attribute__((address_space(1))) v4i_vs*)g,
        (__attribute__((address_space(3))) v4i_vs*)l, 0, 0);
#else
    *(u16x8*)l = *(const u16x8*)g;     // synchronous staging fallback
#endif
}

template <int N> __device__ __forceinline__ void wait_asynccnt() {
#if ASYNC_LDS
#if __has_builtin(__builtin_amdgcn_s_wait_asynccnt)
    __builtin_amdgcn_s_wait_asynccnt(N);
#else
    asm volatile("s_wait_asynccnt %0" ::"i"(N) : "memory");
#endif
#endif
}

// ---------------------------------------------------------------- fp32 -> bf16
__global__ void cvt_bf16_kernel(const float* __restrict__ in,
                                unsigned short* __restrict__ out, int n) {
    int i = blockIdx.x * blockDim.x + threadIdx.x;
    if (i < n) out[i] = f2bf(in[i]);
}

// ------------------------------------------------- C[M,N] = A[M,K] * W[N,K]^T
// A, W bf16 row-major; C fp32 row-major. 4 waves/block, block tile 64x64.
// A/B K-tiles double-buffered in LDS via async-to-LDS copies.
__global__ __launch_bounds__(128)
void gemm_bf16_kernel(const unsigned short* __restrict__ A,
                      const unsigned short* __restrict__ W,
                      float* __restrict__ C, int M, int N, int K) {
    __shared__ unsigned short tA[2][64 * 32];
    __shared__ unsigned short tB[2][64 * 32];
    const int tid  = threadIdx.x;
    const int lane = tid & 31;
    const int wave = tid >> 5;
    const int m0 = blockIdx.y * 64;
    const int n0 = blockIdx.x * 64;
    const int lo = lane & 15;
    const int hi = lane >> 4;

    // stage one 64x32 bf16 tile pair: 256 16B-chunks each, 128 threads x 2
    auto stage = [&](int buf, int kk) {
        #pragma unroll
        for (int c0 = 0; c0 < 2; ++c0) {
            const int c  = tid + c0 * 128;
            const int r  = c >> 2;
            const int co = (c & 3) * 8;
            async_cp16(A + (size_t)(m0 + r) * K + kk + co, &tA[buf][r * 32 + co]);
            async_cp16(W + (size_t)(n0 + r) * K + kk + co, &tB[buf][r * 32 + co]);
        }
    };

    v8f acc[4] = {};
    stage(0, 0);
    int buf = 0;
    for (int k = 0; k < K; k += 32, buf ^= 1) {
        const bool more = (k + 32) < K;
        if (more) stage(buf ^ 1, k + 32);
        if (k + 64 < K) {   // prefetch 2 tiles ahead into L2
            const int c = tid, r = c >> 2, co = (c & 3) * 8;
            __builtin_prefetch(A + (size_t)(m0 + r) * K + k + 64 + co, 0, 1);
            __builtin_prefetch(W + (size_t)(n0 + r) * K + k + 64 + co, 0, 1);
        }
        if (more) wait_asynccnt<4>(); else wait_asynccnt<0>();
        __syncthreads();                     // tile[buf] fully resident

        // A fragment (16x32) from LDS: lanes 0-15 K{0..7,16..23}, 16-31 K{8..15,24..31}
        const unsigned short* ap = &tA[buf][(wave * 16 + lo) * 32 + hi * 8];
        u16x8 a0 = *(const u16x8*)(ap);
        u16x8 a1 = *(const u16x8*)(ap + 16);
        u16x16 af;
        #pragma unroll
        for (int e = 0; e < 8; ++e) { af[e] = a0[e]; af[8 + e] = a1[e]; }
        v16bf av = bfvec(af);
        #pragma unroll
        for (int j = 0; j < 4; ++j) {
            const unsigned short* bp = &tB[buf][(j * 16 + lo) * 32 + hi * 8];
            u16x8 b0 = *(const u16x8*)(bp);
            u16x8 b1 = *(const u16x8*)(bp + 16);
            u16x16 bfr;
            #pragma unroll
            for (int e = 0; e < 8; ++e) { bfr[e] = b0[e]; bfr[8 + e] = b1[e]; }
            acc[j] = __builtin_amdgcn_wmma_f32_16x16x32_bf16(
                false, av, false, bfvec(bfr), (short)0, acc[j], false, false);
        }
        __syncthreads();                     // done reading tile[buf]
    }
    #pragma unroll
    for (int j = 0; j < 4; ++j)
        #pragma unroll
        for (int r = 0; r < 8; ++r)
            C[(size_t)(m0 + wave * 16 + r + hi * 8) * N + n0 + j * 16 + lo] = acc[j][r];
}

// ----------------------------------------- RoPE + bf16 convert, head-major q/k
__global__ void rope_cvt_kernel(const float* __restrict__ qf, const float* __restrict__ kf,
                                const float* __restrict__ rf,
                                unsigned short* __restrict__ qb, unsigned short* __restrict__ kb) {
    const int s = blockIdx.x;
    const int d = threadIdx.x;                 // 0..127
    const float c  = rf[(size_t)s * (HD * 2) + d * 2 + 0];
    const float sn = rf[(size_t)s * (HD * 2) + d * 2 + 1];
    const float sgn = (d & 1) ? 1.0f : -1.0f;  // interleaved adjacent-pair (d, d^1) rotation
    const int dp = d ^ 1;
    for (int h = 0; h < NH; ++h) {
        float x  = qf[(size_t)s * (NH * HD) + h * HD + d];
        float xp = qf[(size_t)s * (NH * HD) + h * HD + dp];
        qb[((size_t)h * SEQ + s) * HD + d] = f2bf(x * c + sgn * xp * sn);
    }
    for (int h = 0; h < NKV; ++h) {
        float x  = kf[(size_t)s * (NKV * HD) + h * HD + d];
        float xp = kf[(size_t)s * (NKV * HD) + h * HD + dp];
        kb[((size_t)h * SEQ + s) * HD + d] = f2bf(x * c + sgn * xp * sn);
    }
}

// ------------------------- V transpose+convert: vf[SEQ][NKV*HD] f32 -> vT[NKV*HD][SEQ] bf16
__global__ __launch_bounds__(256)
void transpose_v_kernel(const float* __restrict__ vf, unsigned short* __restrict__ vT) {
    __shared__ unsigned short tile[32][33];    // pad to avoid bank conflicts
    const int tx = threadIdx.x & 31;
    const int ty = threadIdx.x >> 5;           // 0..7
    #pragma unroll
    for (int i = 0; i < 4; ++i) {
        const int s = blockIdx.x * 32 + ty + i * 8;
        const int d = blockIdx.y * 32 + tx;
        tile[ty + i * 8][tx] = f2bf(vf[(size_t)s * (NKV * HD) + d]);
    }
    __syncthreads();
    #pragma unroll
    for (int i = 0; i < 4; ++i) {
        const int d = blockIdx.y * 32 + ty + i * 8;
        const int s = blockIdx.x * 32 + tx;
        vT[(size_t)d * SEQ + s] = tile[tx][ty + i * 8];
    }
}

// ------------------------------------- flash attention: 1 wave per (head, 16 q-rows)
__global__ __launch_bounds__(32)
void attn_kernel(const unsigned short* __restrict__ qb, const unsigned short* __restrict__ kb,
                 const unsigned short* __restrict__ vT, unsigned short* __restrict__ ab) {
    __shared__ unsigned short Pt[16 * 32];
    const int lane = threadIdx.x & 31;
    const int head = blockIdx.y;
    const int kvh  = head >> 2;                // 32 q-heads / 8 kv-heads
    const int q0   = blockIdx.x * 16;
    const int lo = lane & 15;
    const int hi = lane >> 4;
    const float scale = 0.08838834764831845f;  // 1/sqrt(128)

    const unsigned short* Q   = qb + (size_t)head * SEQ * HD;
    const unsigned short* Km  = kb + (size_t)kvh * SEQ * HD;
    const unsigned short* VmT = vT + (size_t)kvh * HD * SEQ;   // [dim][seq]

    // Q A-fragments: 4 x (16x32) covering head_dim=128
    v16bf qa[4];
    {
        const unsigned short* qp = Q + (size_t)(q0 + lo) * HD;
        #pragma unroll
        for (int i = 0; i < 4; ++i) {
            const unsigned short* p = qp + i * 32 + hi * 8;
            u16x8 c0 = *(const u16x8*)p;
            u16x8 c1 = *(const u16x8*)(p + 16);
            u16x16 t;
            #pragma unroll
            for (int e = 0; e < 8; ++e) { t[e] = c0[e]; t[8 + e] = c1[e]; }
            qa[i] = bfvec(t);
        }
    }

    v8f o[8] = {};
    float mrow[8], lrow[8];
    #pragma unroll
    for (int r = 0; r < 8; ++r) { mrow[r] = -1e30f; lrow[r] = 0.0f; }

    const int nkb = (q0 + 16 + 31) >> 5;       // 32-key blocks covering causal span
    for (int blk = 0; blk < nkb; ++blk) {
        const int kbase = blk * 32;
        v8f s0 = {}, s1 = {};
        #pragma unroll
        for (int i = 0; i < 4; ++i) {
            const unsigned short* kp0 = Km + (size_t)(kbase + lo) * HD + i * 32 + hi * 8;
            u16x8 c0 = *(const u16x8*)kp0;
            u16x8 c1 = *(const u16x8*)(kp0 + 16);
            u16x16 t0;
            #pragma unroll
            for (int e = 0; e < 8; ++e) { t0[e] = c0[e]; t0[8 + e] = c1[e]; }
            s0 = __builtin_amdgcn_wmma_f32_16x16x32_bf16(
                false, qa[i], false, bfvec(t0), (short)0, s0, false, false);
            const unsigned short* kp1 = Km + (size_t)(kbase + 16 + lo) * HD + i * 32 + hi * 8;
            u16x8 d0 = *(const u16x8*)kp1;
            u16x8 d1 = *(const u16x8*)(kp1 + 16);
            u16x16 t1;
            #pragma unroll
            for (int e = 0; e < 8; ++e) { t1[e] = d0[e]; t1[8 + e] = d1[e]; }
            s1 = __builtin_amdgcn_wmma_f32_16x16x32_bf16(
                false, qa[i], false, bfvec(t1), (short)0, s1, false, false);
        }
        // online softmax; row M = r + 8*hi lives in one 16-lane half -> xor 1/2/4/8
        const int key0 = kbase + lo;
        const int key1 = kbase + 16 + lo;
        #pragma unroll
        for (int r = 0; r < 8; ++r) {
            const int qi = q0 + r + hi * 8;
            float v0 = s0[r] * scale; if (key0 > qi) v0 = -1e30f;
            float v1 = s1[r] * scale; if (key1 > qi) v1 = -1e30f;
            float lm = fmaxf(v0, v1);
            lm = fmaxf(lm, __shfl_xor(lm, 1, 32));
            lm = fmaxf(lm, __shfl_xor(lm, 2, 32));
            lm = fmaxf(lm, __shfl_xor(lm, 4, 32));
            lm = fmaxf(lm, __shfl_xor(lm, 8, 32));
            const float mnew  = fmaxf(mrow[r], lm);
            const float alpha = __expf(mrow[r] - mnew);
            const float p0 = __expf(v0 - mnew);
            const float p1 = __expf(v1 - mnew);
            float ps = p0 + p1;
            ps += __shfl_xor(ps, 1, 32);
            ps += __shfl_xor(ps, 2, 32);
            ps += __shfl_xor(ps, 4, 32);
            ps += __shfl_xor(ps, 8, 32);
            lrow[r] = lrow[r] * alpha + ps;
            mrow[r] = mnew;
            #pragma unroll
            for (int n = 0; n < 8; ++n) o[n][r] *= alpha;
            Pt[(r + hi * 8) * 32 + lo]      = f2bf(p0);
            Pt[(r + hi * 8) * 32 + 16 + lo] = f2bf(p1);
        }
        asm volatile("s_wait_dscnt 0x0" ::: "memory");   // LDS in-order per wave
        // reload P in A-matrix layout (16x32)
        u16x16 pt;
        #pragma unroll
        for (int e = 0; e < 8; ++e) {
            pt[e]     = Pt[lo * 32 + hi * 8 + e];
            pt[8 + e] = Pt[lo * 32 + 16 + hi * 8 + e];
        }
        v16bf pv = bfvec(pt);
        // O += P(16x32) @ V(32x128): V^T rows are contiguous along keys
        #pragma unroll
        for (int n = 0; n < 8; ++n) {
            const unsigned short* vp = VmT + (size_t)(n * 16 + lo) * SEQ + kbase + hi * 8;
            u16x8 e0 = *(const u16x8*)vp;
            u16x8 e1 = *(const u16x8*)(vp + 16);
            u16x16 vt;
            #pragma unroll
            for (int e = 0; e < 8; ++e) { vt[e] = e0[e]; vt[8 + e] = e1[e]; }
            o[n] = __builtin_amdgcn_wmma_f32_16x16x32_bf16(
                false, pv, false, bfvec(vt), (short)0, o[n], false, false);
        }
    }
    // normalize; write bf16 attn output in [seq][head*HD] layout for final GEMM
    #pragma unroll
    for (int r = 0; r < 8; ++r) {
        const float inv = 1.0f / lrow[r];
        const int qi = q0 + r + hi * 8;
        #pragma unroll
        for (int n = 0; n < 8; ++n)
            ab[(size_t)qi * DIM + head * HD + n * 16 + lo] = f2bf(o[n][r] * inv);
    }
}

// -----------------------------------------------------------------------------
extern "C" void kernel_launch(void* const* d_in, const int* in_sizes, int n_in,
                              void* d_out, int out_size, void* d_ws, size_t ws_size,
                              hipStream_t stream) {
    (void)in_sizes; (void)n_in; (void)out_size; (void)ws_size;
    const float* x  = (const float*)d_in[0];
    const float* wq = (const float*)d_in[1];
    const float* wk = (const float*)d_in[2];
    const float* wv = (const float*)d_in[3];
    const float* wo = (const float*)d_in[4];
    const float* rf = (const float*)d_in[5];
    // d_in[6] = start_pos (0 in reference; positions used absolutely)

    char* p = (char*)d_ws;
    auto alloc_us = [&](size_t n) { unsigned short* r = (unsigned short*)p; p += n * 2; return r; };
    auto alloc_f  = [&](size_t n) { float* r = (float*)p; p += n * 4; return r; };

    unsigned short* xb  = alloc_us((size_t)SEQ * DIM);
    unsigned short* wqb = alloc_us((size_t)DIM * DIM);
    unsigned short* wkb = alloc_us((size_t)NKV * HD * DIM);
    unsigned short* wvb = alloc_us((size_t)NKV * HD * DIM);
    unsigned short* wob = alloc_us((size_t)DIM * DIM);
    float* qf = alloc_f((size_t)SEQ * DIM);
    float* kf = alloc_f((size_t)SEQ * NKV * HD);
    float* vf = alloc_f((size_t)SEQ * NKV * HD);
    unsigned short* qbf = alloc_us((size_t)SEQ * DIM);
    unsigned short* kbf = alloc_us((size_t)SEQ * NKV * HD);
    unsigned short* vTb = alloc_us((size_t)NKV * HD * SEQ);
    unsigned short* ab  = alloc_us((size_t)SEQ * DIM);

    auto cvt = [&](const float* in, unsigned short* out, size_t n) {
        cvt_bf16_kernel<<<dim3((unsigned)(n / 256)), 256, 0, stream>>>(in, out, (int)n);
    };
    cvt(x,  xb,  (size_t)SEQ * DIM);
    cvt(wq, wqb, (size_t)DIM * DIM);
    cvt(wk, wkb, (size_t)NKV * HD * DIM);
    cvt(wv, wvb, (size_t)NKV * HD * DIM);
    cvt(wo, wob, (size_t)DIM * DIM);

    gemm_bf16_kernel<<<dim3(DIM / 64,      SEQ / 64), 128, 0, stream>>>(xb, wqb, qf, SEQ, DIM,      DIM);
    gemm_bf16_kernel<<<dim3(NKV * HD / 64, SEQ / 64), 128, 0, stream>>>(xb, wkb, kf, SEQ, NKV * HD, DIM);
    gemm_bf16_kernel<<<dim3(NKV * HD / 64, SEQ / 64), 128, 0, stream>>>(xb, wvb, vf, SEQ, NKV * HD, DIM);

    rope_cvt_kernel<<<SEQ, HD, 0, stream>>>(qf, kf, rf, qbf, kbf);
    transpose_v_kernel<<<dim3(SEQ / 32, NKV * HD / 32), 256, 0, stream>>>(vf, vTb);

    attn_kernel<<<dim3(SEQ / 16, NH), 32, 0, stream>>>(qbf, kbf, vTb, ab);

    gemm_bf16_kernel<<<dim3(DIM / 64, SEQ / 64), 128, 0, stream>>>(ab, wob, (float*)d_out, SEQ, DIM, DIM);
}
